// MaskRCNN_88373247083257
// MI455X (gfx1250) — matrix operand order repaired and checked by reading, not compiled, not run on hardware
//
#include <hip/hip_runtime.h>
#include <hip/hip_bf16.h>
#include <cstdint>
#include <cstddef>

typedef __attribute__((ext_vector_type(16))) _Float16 v16h;
typedef __attribute__((ext_vector_type(8)))  float    v8f;

#define NIMG        2
#define PPOS        87296          /* positions per image (256^2+128^2+64^2+32^2+16^2) */
#define NPOS        174592         /* NIMG * PPOS */
#define NANCH       261888         /* PPOS * 3 anchors */
#define CIN         256
#define CMID        512
#define KSTEPS      72             /* 9 taps * (256/32) */
#define MTILES      10912          /* NPOS / 16 */
#define NTILES      32             /* 512 / 16 */
#define PRE_NMS     6000
#define POST_NMS    2000
#define SELCAP      8192
#define NBINS       4096
#define NMS_THR     0.7f

// ---------------------------------------------------------------------------
// 1) Pack shared-conv weights (HWIO f32) into per-lane WMMA B fragments (f16)
//    layout: Bw[ks][ocTile][lane][16], ks = tap*8 + icChunk
// ---------------------------------------------------------------------------
__global__ void rpn_prep_weights(const float* __restrict__ w, _Float16* __restrict__ Bw) {
    int t = blockIdx.x * blockDim.x + threadIdx.x;       // t < KSTEPS*NTILES*32
    if (t >= KSTEPS * NTILES * 32) return;
    int lane   = t & 31;
    int ocTile = (t >> 5) & 31;
    int ks     = t >> 10;
    int tap    = ks >> 3;
    int ic0    = (ks & 7) * 32;
    int oc     = ocTile * 16 + (lane & 15);
    int khalf  = (lane < 16) ? 0 : 16;                   // B: lanes 0-15 K=0..15, 16-31 K=16..31
    _Float16* dst = Bw + (size_t)t * 16;
    #pragma unroll
    for (int j = 0; j < 16; ++j) {
        int ic = ic0 + khalf + j;
        dst[j] = (_Float16)w[((size_t)tap * CIN + ic) * CMID + oc];
    }
}

// ---------------------------------------------------------------------------
// 2) Zero/seed the selection scratch (heads no longer need global init)
// ---------------------------------------------------------------------------
__global__ void rpn_init(int* __restrict__ hist, int* __restrict__ cnt, int* __restrict__ thr,
                         float* __restrict__ selScore, int* __restrict__ selIdx) {
    int t = blockIdx.x * blockDim.x + threadIdx.x;
    if (t < NIMG * NBINS) hist[t] = 0;
    if (t < NIMG) { cnt[t] = 0; thr[t] = 0; }
    if (t < NIMG * SELCAP) { selScore[t] = -1.0e30f; selIdx[t] = -1; }
}

// ---------------------------------------------------------------------------
// 3) Shared 3x3 conv (256->512) + ReLU as implicit GEMM on WMMA.
//    One block = 256 threads = 8 waves; each wave owns 4 N-tiles (4 WMMAs per
//    K-step), so one block covers all 512 output channels of a 16-position
//    M-tile. A tile staged cooperatively in LDS (each thread: one float2 load
//    + 2 f16 converts). The 1x1 heads reduce in LDS and write directly.
// ---------------------------------------------------------------------------
__global__ __launch_bounds__(256) void rpn_conv_wmma(
    const float* __restrict__ p2, const float* __restrict__ p3,
    const float* __restrict__ p4, const float* __restrict__ p5,
    const float* __restrict__ p6,
    const _Float16* __restrict__ Bw,
    const float* __restrict__ b_shared,
    const float* __restrict__ w_class, const float* __restrict__ b_class,
    const float* __restrict__ w_bbox,  const float* __restrict__ b_bbox,
    float* __restrict__ logits, float* __restrict__ deltas)
{
    const int offs[6]   = {0, 65536, 81920, 86016, 87040, 87296};
    const int widths[5] = {256, 128, 64, 32, 16};
    const float* feats[5] = {p2, p3, p4, p5, p6};

    __builtin_amdgcn_s_wait_tensorcnt(0);                // gfx1250 path (trivially satisfied)

    const int tid   = threadIdx.x;
    const int wave  = tid >> 5;
    const int lane  = tid & 31;
    const int mTile = blockIdx.x;
    const int nTileBase = wave << 2;                     // 4 ocTiles per wave

    // ---- A-stage mapping: thread t fills halves e=2t, 2t+1 of ldsA[512]
    //      element (laneE, j): M = laneE&15, K = kbase + (laneE<16 ? 0 : 16) + j
    const int e     = tid << 1;
    const int laneE = e >> 4;
    const int j0    = e & 15;                            // even
    const int mE    = laneE & 15;
    const int kE    = ((laneE < 16) ? 0 : 16) + j0;

    // decode output position for row mE (done once; reused every K-step)
    const int pmE = mTile * 16 + mE;
    const int nE  = pmE / PPOS;
    int pE = pmE - nE * PPOS;
    int lvlE = 0;
    while (pE >= offs[lvlE + 1]) ++lvlE;
    const int WE   = widths[lvlE];
    const int relE = pE - offs[lvlE];
    const int yE   = relE / WE;
    const int xE   = relE - yE * WE;
    const float* fbaseE = feats[lvlE] + (size_t)nE * WE * WE * CIN;

    __shared__ _Float16 ldsA[16 * 32];                   // one 16x32 f16 A tile
    __shared__ float accLDS[16 * 18];                    // per-position head partials

    v8f acc0 = {}, acc1 = {}, acc2 = {}, acc3 = {};

    for (int ks = 0; ks < KSTEPS; ++ks) {
        const int tap = ks >> 3;
        const int ic0 = (ks & 7) * 32;
        __syncthreads();                                 // prior A reads complete
        {
            const int yy = yE + tap / 3 - 1;
            const int xx = xE + tap % 3 - 1;
            float vx = 0.0f, vy = 0.0f;
            if (yy >= 0 && yy < WE && xx >= 0 && xx < WE) {
                const float* s = fbaseE + (size_t)(yy * WE + xx) * CIN + ic0 + kE;
                const float2 v = *(const float2*)s;
                vx = v.x; vy = v.y;
                __builtin_prefetch(s + 32, 0, 0);        // next ic chunk -> global_prefetch_b8
            }
            ldsA[e]     = (_Float16)vx;
            ldsA[e + 1] = (_Float16)vy;
        }
        __syncthreads();                                 // A tile staged

        const v16h a = *(const v16h*)(ldsA + (lane << 4));            // ds_load_b128 x2
        const _Float16* bb = Bw + (((size_t)ks * NTILES + nTileBase) * 32 + lane) * 16;
        const v16h b0 = *(const v16h*)(bb);
        const v16h b1 = *(const v16h*)(bb + 512);
        const v16h b2 = *(const v16h*)(bb + 1024);
        const v16h b3 = *(const v16h*)(bb + 1536);
        acc0 = __builtin_amdgcn_wmma_f32_16x16x32_f16(false, a, false, b0, (short)0, acc0, false, false);
        acc1 = __builtin_amdgcn_wmma_f32_16x16x32_f16(false, a, false, b1, (short)0, acc1, false, false);
        acc2 = __builtin_amdgcn_wmma_f32_16x16x32_f16(false, a, false, b2, (short)0, acc2, false, false);
        acc3 = __builtin_amdgcn_wmma_f32_16x16x32_f16(false, a, false, b3, (short)0, acc3, false, false);
    }

    // ---- Epilogue: bias + ReLU, fold the 1x1 heads across all 512 channels
    for (int i = tid; i < 16 * 18; i += 256) accLDS[i] = 0.0f;
    __syncthreads();

    const int ncol = lane & 15;                          // C/D layout: N = lane & 15
    const int mrow = (lane >> 4) << 3;                   // rows r / r+8
    #pragma unroll
    for (int u = 0; u < 4; ++u) {
        const int oc = (nTileBase + u) * 16 + ncol;
        const float bias = b_shared[oc];
        float wcl[6], wbl[12];
        #pragma unroll
        for (int c = 0; c < 6;  ++c) wcl[c] = w_class[oc * 6 + c];
        #pragma unroll
        for (int c = 0; c < 12; ++c) wbl[c] = w_bbox[oc * 12 + c];
        const v8f accu = (u == 0) ? acc0 : (u == 1) ? acc1 : (u == 2) ? acc2 : acc3;
        #pragma unroll
        for (int r = 0; r < 8; ++r) {
            const float v = fmaxf(accu[r] + bias, 0.0f);
            const int m = mrow + r;
            #pragma unroll
            for (int c = 0; c < 6;  ++c) atomicAdd(&accLDS[m * 18 + c], v * wcl[c]);
            #pragma unroll
            for (int c = 0; c < 12; ++c) atomicAdd(&accLDS[m * 18 + 6 + c], v * wbl[c]);
        }
    }
    __syncthreads();
    for (int i = tid; i < 16 * 18; i += 256) {
        const int m = i / 18, c = i - m * 18;
        const size_t pg = (size_t)mTile * 16 + m;        // one block per M-tile: plain stores
        if (c < 6) logits[pg * 6 + c]        = b_class[c]      + accLDS[i];
        else       deltas[pg * 12 + (c - 6)] = b_bbox[c - 6]   + accLDS[i];
    }
}

// ---------------------------------------------------------------------------
// 4) Softmax -> foreground score per anchor
// ---------------------------------------------------------------------------
__global__ void rpn_scores(const float* __restrict__ logits, float* __restrict__ scores) {
    int t = blockIdx.x * blockDim.x + threadIdx.x;
    if (t >= NIMG * NANCH) return;
    int n = t / NANCH, r = t - n * NANCH;
    int p = r / 3, a = r - p * 3;
    size_t base = ((size_t)n * PPOS + p) * 6 + a * 2;
    float l0 = logits[base], l1 = logits[base + 1];
    scores[t] = 1.0f / (1.0f + __expf(l0 - l1));
}

// 5) score histogram (per image)
__global__ void rpn_hist(const float* __restrict__ scores, int* __restrict__ hist) {
    int t = blockIdx.x * blockDim.x + threadIdx.x;
    if (t >= NIMG * NANCH) return;
    int n = t / NANCH;
    int bin = (int)(scores[t] * (float)NBINS);
    bin = min(max(bin, 0), NBINS - 1);
    atomicAdd(&hist[n * NBINS + bin], 1);
}

// 6) per-image threshold bin such that count(bin >= thr) >= PRE_NMS
__global__ void rpn_thresh(const int* __restrict__ hist, int* __restrict__ thr) {
    int n = blockIdx.x;
    if (threadIdx.x != 0) return;
    int acc = 0, t = 0;
    for (int b = NBINS - 1; b >= 0; --b) {
        acc += hist[n * NBINS + b];
        if (acc >= PRE_NMS) { t = b; break; }
    }
    thr[n] = t;
}

// 7) compact candidates above threshold
__global__ void rpn_compact(const float* __restrict__ scores, const int* __restrict__ thr,
                            int* __restrict__ cnt, float* __restrict__ selScore,
                            int* __restrict__ selIdx) {
    int t = blockIdx.x * blockDim.x + threadIdx.x;
    if (t >= NIMG * NANCH) return;
    int n = t / NANCH, r = t - n * NANCH;
    float s = scores[t];
    int bin = (int)(s * (float)NBINS);
    bin = min(max(bin, 0), NBINS - 1);
    if (bin >= thr[n]) {
        int slot = atomicAdd(&cnt[n], 1);
        if (slot < SELCAP) {
            selScore[(size_t)n * SELCAP + slot] = s;
            selIdx[(size_t)n * SELCAP + slot]   = r;
        }
    }
}

// 8) per-image bitonic sort (descending by score), SELCAP elements, one block
__global__ __launch_bounds__(1024) void rpn_sort(float* __restrict__ selScore,
                                                 int* __restrict__ selIdx) {
    const int n = blockIdx.x;
    float* s = selScore + (size_t)n * SELCAP;
    int*  id = selIdx   + (size_t)n * SELCAP;
    for (int k = 2; k <= SELCAP; k <<= 1) {
        for (int j = k >> 1; j > 0; j >>= 1) {
            __syncthreads();
            for (int i = threadIdx.x; i < SELCAP; i += 1024) {
                int q = i ^ j;
                if (q > i) {
                    bool desc = ((i & k) == 0);
                    float si = s[i], sq = s[q];
                    bool sw = desc ? (si < sq) : (si > sq);
                    if (sw) {
                        s[i] = sq; s[q] = si;
                        int tmp = id[i]; id[i] = id[q]; id[q] = tmp;
                    }
                }
            }
        }
    }
}

// 9) decode top PRE_NMS proposals: anchors + std*deltas, exp, clip
__global__ void rpn_propose(const float* __restrict__ anchors, const float* __restrict__ deltas,
                            const int* __restrict__ selIdx, float* __restrict__ boxes) {
    int t = blockIdx.x * blockDim.x + threadIdx.x;
    if (t >= NIMG * PRE_NMS) return;
    int n = t / PRE_NMS, i = t - n * PRE_NMS;
    int idx = selIdx[(size_t)n * SELCAP + i];
    float o0 = 0.f, o1 = 0.f, o2 = 0.f, o3 = 0.f;
    if (idx >= 0) {
        int p = idx / 3, a = idx - p * 3;
        const float* A = anchors + (size_t)idx * 4;
        size_t db = ((size_t)n * PPOS + p) * 12 + a * 4;
        float d0 = deltas[db + 0] * 0.1f, d1 = deltas[db + 1] * 0.1f;
        float d2 = deltas[db + 2] * 0.2f, d3 = deltas[db + 3] * 0.2f;
        float h = A[2] - A[0], w = A[3] - A[1];
        float cy = A[0] + 0.5f * h + d0 * h;
        float cx = A[1] + 0.5f * w + d1 * w;
        h *= __expf(d2); w *= __expf(d3);
        o0 = fminf(fmaxf(cy - 0.5f * h, 0.f), 1.f);
        o1 = fminf(fmaxf(cx - 0.5f * w, 0.f), 1.f);
        o2 = fminf(fmaxf(cy + 0.5f * h, 0.f), 1.f);
        o3 = fminf(fmaxf(cx + 0.5f * w, 0.f), 1.f);
    }
    float* B = boxes + (size_t)t * 4;
    B[0] = o0; B[1] = o1; B[2] = o2; B[3] = o3;
}

// 10) sequential greedy NMS, one block per image, keep flags in LDS
__global__ __launch_bounds__(1024) void rpn_nms(const float* __restrict__ boxes,
                                                int* __restrict__ keepOut) {
    const int n = blockIdx.x;
    const float* B = boxes + (size_t)n * PRE_NMS * 4;
    __shared__ unsigned char keep[PRE_NMS];
    __shared__ int anyflag;
    for (int i = threadIdx.x; i < PRE_NMS; i += 1024) keep[i] = 0;
    __syncthreads();
    for (int i = 0; i < PRE_NMS; ++i) {
        float iy1 = B[i * 4 + 0], ix1 = B[i * 4 + 1], iy2 = B[i * 4 + 2], ix2 = B[i * 4 + 3];
        float areaI = (iy2 - iy1) * (ix2 - ix1);
        if (threadIdx.x == 0) anyflag = 0;
        __syncthreads();
        int local = 0;
        for (int j = threadIdx.x; j < i; j += 1024) {
            if (keep[j]) {
                float y1 = fmaxf(iy1, B[j * 4 + 0]);
                float x1 = fmaxf(ix1, B[j * 4 + 1]);
                float y2 = fminf(iy2, B[j * 4 + 2]);
                float x2 = fminf(ix2, B[j * 4 + 3]);
                float inter = fmaxf(y2 - y1, 0.f) * fmaxf(x2 - x1, 0.f);
                float areaJ = (B[j * 4 + 2] - B[j * 4 + 0]) * (B[j * 4 + 3] - B[j * 4 + 1]);
                float iou = inter / (areaI + areaJ - inter + 1e-8f);
                if (iou > NMS_THR) { local = 1; break; }
            }
        }
        if (local) atomicOr(&anyflag, 1);
        __syncthreads();
        if (threadIdx.x == 0) keep[i] = anyflag ? 0 : 1;
        __syncthreads();
    }
    for (int i = threadIdx.x; i < PRE_NMS; i += 1024) keepOut[n * PRE_NMS + i] = keep[i];
}

// 11) compact kept boxes into the POST_NMS output rows, zero-fill the rest
__global__ void rpn_finalize(const float* __restrict__ boxes, const int* __restrict__ keep,
                             float* __restrict__ out) {
    const int n = blockIdx.x;
    if (threadIdx.x != 0) return;
    int cnt = 0;
    for (int i = 0; i < PRE_NMS && cnt < POST_NMS; ++i) {
        if (keep[n * PRE_NMS + i]) {
            for (int c = 0; c < 4; ++c)
                out[((size_t)n * POST_NMS + cnt) * 4 + c] = boxes[((size_t)n * PRE_NMS + i) * 4 + c];
            ++cnt;
        }
    }
    for (; cnt < POST_NMS; ++cnt)
        for (int c = 0; c < 4; ++c)
            out[((size_t)n * POST_NMS + cnt) * 4 + c] = 0.0f;
}

// ---------------------------------------------------------------------------
extern "C" void kernel_launch(void* const* d_in, const int* in_sizes, int n_in,
                              void* d_out, int out_size, void* d_ws, size_t ws_size,
                              hipStream_t stream) {
    (void)in_sizes; (void)n_in; (void)out_size; (void)ws_size;
    const float* p2      = (const float*)d_in[0];
    const float* p3      = (const float*)d_in[1];
    const float* p4      = (const float*)d_in[2];
    const float* p5      = (const float*)d_in[3];
    const float* p6      = (const float*)d_in[4];
    const float* anchors = (const float*)d_in[5];
    const float* w_sh    = (const float*)d_in[6];
    const float* b_sh    = (const float*)d_in[7];
    const float* w_cls   = (const float*)d_in[8];
    const float* b_cls   = (const float*)d_in[9];
    const float* w_bbx   = (const float*)d_in[10];
    const float* b_bbx   = (const float*)d_in[11];
    float* out = (float*)d_out;

    // carve workspace
    auto up = [](size_t x) { return (x + 255) & ~(size_t)255; };
    char* w = (char*)d_ws;
    _Float16* Bw    = (_Float16*)w; w += up((size_t)KSTEPS * NTILES * 32 * 16 * sizeof(_Float16));
    float* logits   = (float*)w;    w += up((size_t)NPOS * 6 * sizeof(float));
    float* deltas   = (float*)w;    w += up((size_t)NPOS * 12 * sizeof(float));
    float* scores   = (float*)w;    w += up((size_t)NIMG * NANCH * sizeof(float));
    int*   hist     = (int*)w;      w += up((size_t)NIMG * NBINS * sizeof(int));
    int*   thr      = (int*)w;      w += up((size_t)NIMG * sizeof(int));
    int*   cnt      = (int*)w;      w += up((size_t)NIMG * sizeof(int));
    float* selScore = (float*)w;    w += up((size_t)NIMG * SELCAP * sizeof(float));
    int*   selIdx   = (int*)w;      w += up((size_t)NIMG * SELCAP * sizeof(int));
    float* boxes    = (float*)w;    w += up((size_t)NIMG * PRE_NMS * 4 * sizeof(float));
    int*   keep     = (int*)w;      w += up((size_t)NIMG * PRE_NMS * sizeof(int));

    rpn_prep_weights<<<(KSTEPS * NTILES * 32 + 255) / 256, 256, 0, stream>>>(w_sh, Bw);
    rpn_init<<<(NIMG * SELCAP + 255) / 256, 256, 0, stream>>>(hist, cnt, thr, selScore, selIdx);
    rpn_conv_wmma<<<MTILES, 256, 0, stream>>>(p2, p3, p4, p5, p6, Bw, b_sh,
                                              w_cls, b_cls, w_bbx, b_bbx, logits, deltas);
    rpn_scores<<<(NIMG * NANCH + 255) / 256, 256, 0, stream>>>(logits, scores);
    rpn_hist<<<(NIMG * NANCH + 255) / 256, 256, 0, stream>>>(scores, hist);
    rpn_thresh<<<NIMG, 32, 0, stream>>>(hist, thr);
    rpn_compact<<<(NIMG * NANCH + 255) / 256, 256, 0, stream>>>(scores, thr, cnt,
                                                                selScore, selIdx);
    rpn_sort<<<NIMG, 1024, 0, stream>>>(selScore, selIdx);
    rpn_propose<<<(NIMG * PRE_NMS + 255) / 256, 256, 0, stream>>>(anchors, deltas,
                                                                  selIdx, boxes);
    rpn_nms<<<NIMG, 1024, 0, stream>>>(boxes, keep);
    rpn_finalize<<<NIMG, 32, 0, stream>>>(boxes, keep, out);
}
